// LocalCorrLayer_18373870092923
// MI455X (gfx1250) — matrix operand order repaired and checked by reading, not compiled, not run on hardware
//
#include <hip/hip_runtime.h>
#include <hip/hip_bf16.h>

// ---------------------------------------------------------------------------
// Local correlation layer (cost volume), MAX_DISP=4, via f16 WMMA on gfx1250.
// Phase 1: L2-normalize over channels -> f16, pixel-major [b][y][x][c].
//          (padded fqn buffer: zeros in the halo, matching the reference.)
// Phase 2: banded 16x16 cross-correlation per (b, y, x-tile, dy) using
//          V_WMMA_F32_16X16X32_F16, diagonal extraction through LDS.
// ---------------------------------------------------------------------------

#define BB 8
#define CC 256
#define HH 128
#define WW 160
#define RR 4
#define HP (HH + 2 * RR)   // 136
#define WP (WW + 2 * RR)   // 168
#define ND 81              // (2r+1)^2

typedef __attribute__((ext_vector_type(16))) _Float16 v16h;
typedef __attribute__((ext_vector_type(8)))  _Float16 v8h;
typedef __attribute__((ext_vector_type(8)))  float    v8f;

// ---------------------------------------------------------------------------
// Phase 1a: normalize fr over channels, f32 NCHW -> f16 pixel-major [b][y][x][c]
// One thread per pixel; channel loads are coalesced across adjacent lanes.
// ---------------------------------------------------------------------------
__global__ __launch_bounds__(256) void lc_norm_fr(const float* __restrict__ fr,
                                                  _Float16* __restrict__ frn) {
    const int HW = HH * WW;
    const int t = blockIdx.x * 256 + threadIdx.x;      // 0 .. B*H*W-1 (exact)
    const int b = t / HW;
    const int p = t % HW;
    const float* src = fr + (size_t)b * CC * HW + p;

    float s = 0.0f;
#pragma unroll 8
    for (int c = 0; c < CC; ++c) {
        float v = src[(size_t)c * HW];
        s += v * v;
    }
    const float scale = 1.0f / fmaxf(sqrtf(s), 1e-12f);

    _Float16* dst = frn + (size_t)t * CC;
    for (int c0 = 0; c0 < CC; c0 += 8) {
        v8h v;
#pragma unroll
        for (int j = 0; j < 8; ++j)
            v[j] = (_Float16)(src[(size_t)(c0 + j) * HW] * scale);
        *(v8h*)(dst + c0) = v;
    }
}

// ---------------------------------------------------------------------------
// Phase 1b: normalize zero-padded fq -> f16 pixel-major [b][yp][xp][c]
// ---------------------------------------------------------------------------
__global__ __launch_bounds__(256) void lc_norm_fq(const float* __restrict__ fq,
                                                  _Float16* __restrict__ fqn) {
    const int HW = HH * WW;
    const int t = blockIdx.x * 256 + threadIdx.x;      // 0 .. B*HP*WP-1 (exact)
    const int b   = t / (HP * WP);
    const int rem = t % (HP * WP);
    const int yp  = rem / WP;
    const int xp  = rem % WP;

    _Float16* dst = fqn + (size_t)t * CC;
    const bool pad = (yp < RR) | (yp >= HH + RR) | (xp < RR) | (xp >= WW + RR);
    if (pad) {
        for (int c0 = 0; c0 < CC; c0 += 8) {
            v8h z = {};
            *(v8h*)(dst + c0) = z;
        }
        return;
    }
    const float* src = fq + (size_t)b * CC * HW + (size_t)(yp - RR) * WW + (xp - RR);

    float s = 0.0f;
#pragma unroll 8
    for (int c = 0; c < CC; ++c) {
        float v = src[(size_t)c * HW];
        s += v * v;
    }
    const float scale = 1.0f / fmaxf(sqrtf(s), 1e-12f);

    for (int c0 = 0; c0 < CC; c0 += 8) {
        v8h v;
#pragma unroll
        for (int j = 0; j < 8; ++j)
            v[j] = (_Float16)(src[(size_t)(c0 + j) * HW] * scale);
        *(v8h*)(dst + c0) = v;
    }
}

// ---------------------------------------------------------------------------
// Phase 2: correlation.
// Grid: (W/16, H, B). Block: 288 threads = 9 waves; wave w handles dy = w.
// Each wave: D(m,n) = sum_c frn(c, x0+m) * fqn_pad(c, y+dy, x0+n), n in [0,32),
// via 2 column-tiles x 8 K-chunks of v_wmma_f32_16x16x32_f16.
// Wanted outputs are the diagonals n = m + dx, dx in [0,9).
// ---------------------------------------------------------------------------
__global__ __launch_bounds__(288) void lc_corr(const _Float16* __restrict__ frn,
                                               const _Float16* __restrict__ fqn,
                                               float* __restrict__ out) {
    __shared__ float lds[9 * 16 * 34];                 // per-wave 16x32 (+2 pad) f32 tile

    const int lane = threadIdx.x & 31;
    const int dy   = threadIdx.x >> 5;                 // 0..8
    const int m    = lane & 15;                        // A row / B column / C column
    const int hi   = lane >> 4;                        // half-wave select
    const int x0   = blockIdx.x * 16;
    const int y    = blockIdx.y;
    const int b    = blockIdx.z;

    // ---- A fragments: 16x32 f16 per chunk, contiguous K order (matches B):
    // lanes 0-15: K = k0+{0..15}; lanes 16-31: K = k0+{16..31}
    const _Float16* aptr = frn + ((size_t)(b * HH + y) * WW + x0 + m) * CC;
    v16h a[8];
#pragma unroll
    for (int kc = 0; kc < 8; ++kc) {
        const int k0 = kc * 32;
        a[kc] = *(const v16h*)(aptr + k0 + hi * 16);
    }

    // ---- B fragments + accumulation. B layout: lane n holds column n,
    // K = k0 + (hi ? 16 : 0) + [0..15] as 16 consecutive f16 (one 32B load).
    const _Float16* brow = fqn + (size_t)(b * HP + y + dy) * WP * CC;
    float* lbase = &lds[dy * 16 * 34];

#pragma unroll
    for (int t = 0; t < 2; ++t) {
        int px = x0 + 16 * t + m;                      // padded-x column index
        px = (px > WP - 1) ? (WP - 1) : px;            // branchless clamp (cols unused)
        const _Float16* bptr = brow + (size_t)px * CC + hi * 16;

        v8f acc = {};
#pragma unroll
        for (int kc = 0; kc < 8; ++kc) {
            v16h bf = *(const v16h*)(bptr + kc * 32);
            acc = __builtin_amdgcn_wmma_f32_16x16x32_f16(
                false, a[kc], false, bf, (short)0, acc, false, false);
        }
        // C/D layout: VGPR r -> row (r + 8*hi), column = lane&15.
#pragma unroll
        for (int r = 0; r < 8; ++r)
            lbase[(r + hi * 8) * 34 + t * 16 + m] = acc[r];
    }

    __syncthreads();

    // ---- Extract diagonals: out[b, dy*9+dx, y, x0+m] = D(m, m+dx) / C
    const float* lrow = &lds[dy * 16 * 34 + m * 34];
    float* orow = out + (((size_t)b * ND + dy * 9) * HH + y) * WW + x0 + m;
#pragma unroll
    for (int i = 0; i < 5; ++i) {
        const int dx = 2 * i + hi;
        if (dx < 9)
            orow[(size_t)dx * (HH * WW)] = lrow[m + dx] * (1.0f / 256.0f);
    }
}

// ---------------------------------------------------------------------------
extern "C" void kernel_launch(void* const* d_in, const int* in_sizes, int n_in,
                              void* d_out, int out_size, void* d_ws, size_t ws_size,
                              hipStream_t stream) {
    const float* fr = (const float*)d_in[0];
    const float* fq = (const float*)d_in[1];
    float* out = (float*)d_out;

    // Workspace: frn (f16, B*H*W*C) then fqn (f16, B*HP*WP*C). ~177.5 MB total.
    _Float16* frn = (_Float16*)d_ws;
    _Float16* fqn = frn + (size_t)BB * HH * WW * CC;

    lc_norm_fr<<<(BB * HH * WW) / 256, 256, 0, stream>>>(fr, frn);
    lc_norm_fq<<<(BB * HP * WP + 255) / 256, 256, 0, stream>>>(fq, fqn);

    dim3 grid(WW / 16, HH, BB);
    lc_corr<<<grid, 288, 0, stream>>>(frn, fqn, out);
}
